// GraphormerLayer_17308718203039
// MI455X (gfx1250) — matrix operand order repaired and checked by reading, not compiled, not run on hardware
//
#include <hip/hip_runtime.h>
#include <cstdint>
#include <cstddef>

// ---------------------------------------------------------------------------
// Graphormer layer for MI455X (gfx1250, wave32, WMMA bf16 16x16x32).
// D=1024, H=16 (hd=64), G=32 graphs x N=512 nodes, T=16384.
// GEMMs use double-buffered LDS fed by CDNA5 async global->LDS loads.
// ---------------------------------------------------------------------------

typedef __attribute__((ext_vector_type(16))) __bf16 v16bf;
typedef __attribute__((ext_vector_type(8)))  float  v8f;
typedef __attribute__((ext_vector_type(4)))  int    v4i;

#define DD      1024
#define HH      16
#define HD      64
#define GG      32
#define NN      512
#define TT      (GG * NN)
#define TRIPLE  (3 * DD)

union FragAB {           // 16x32 bf16 A-frag or 32x16 bf16 B-frag (8 VGPRs)
  v16bf v;
  __bf16 e[16];
  uint4  q[2];
};
union FragC {            // 16x16 f32 C/D (8 VGPRs)
  v8f   v;
  float e[8];
};

__device__ inline v8f wmma_bf16(const v16bf& a, const v16bf& b, const v8f& c) {
  return __builtin_amdgcn_wmma_f32_16x16x32_bf16(
      /*neg_a=*/false, a, /*neg_b=*/false, b,
      /*c_mod=*/(short)0, c, /*reuse_a=*/false, /*reuse_b=*/false);
}

// ---- CDNA5 async global->LDS (ASYNCcnt) with inline-asm fallback ----------
typedef __attribute__((address_space(1))) v4i* gv4i_p;  // global int4*
typedef __attribute__((address_space(3))) v4i* lv4i_p;  // LDS int4*

__device__ inline void async_load_b128(const __bf16* g, __bf16* l) {
#if __has_builtin(__builtin_amdgcn_global_load_async_to_lds_b128)
  __builtin_amdgcn_global_load_async_to_lds_b128(
      (gv4i_p)(v4i*)(void*)g,
      (lv4i_p)(v4i*)(void*)l,
      /*imm offset=*/0, /*cpol=*/0);
#else
  asm volatile("global_load_async_to_lds_b128 %0, %1, off"
               :: "v"((uint32_t)(uintptr_t)l), "v"(g) : "memory");
#endif
}

__device__ inline void wait_async0() {
#if __has_builtin(__builtin_amdgcn_s_wait_asynccnt)
  __builtin_amdgcn_s_wait_asynccnt(0);
#else
  asm volatile("s_wait_asynccnt 0x0" ::: "memory");
#endif
}

// ---------------------------------------------------------------------------
// f32 -> bf16 convert
// ---------------------------------------------------------------------------
__global__ __launch_bounds__(256)
void cvt_f32_bf16(const float* __restrict__ in, __bf16* __restrict__ out, int n) {
  int i = blockIdx.x * 256 + threadIdx.x;
  if (i < n) out[i] = (__bf16)in[i];
}

// ---------------------------------------------------------------------------
// C[M,N] = A[M,K](bf16,row) @ B[N,K](bf16,row)^T + bias, optional ReLU,
// output f32 or bf16.  Block tile 128x128, BK=32, 256 threads = 8 waves,
// wave grid 2(M) x 4(N), each wave 64x32 = 4x2 WMMA tiles.
// Double-buffered LDS, tiles streamed with async global->LDS B128 loads.
// ---------------------------------------------------------------------------
template <bool OUT_BF16, bool RELU>
__global__ __launch_bounds__(256)
void gemm_bf16_kernel(const __bf16* __restrict__ A,
                      const __bf16* __restrict__ B,
                      const float*  __restrict__ bias,
                      void* __restrict__ Cout,
                      int M, int N, int K) {
  constexpr int BM = 128, BN = 128, BK = 32, LDST = BK + 8; // elem stride
  __shared__ __attribute__((aligned(16))) __bf16 As[2][BM * LDST];
  __shared__ __attribute__((aligned(16))) __bf16 Bs[2][BN * LDST];

  const int bm = blockIdx.y * BM;
  const int bn = blockIdx.x * BN;
  const int t    = threadIdx.x;
  const int lane = t & 31;
  const int wv   = t >> 5;
  const int wm   = (wv & 1) * 64;   // wave M offset inside block
  const int wn   = (wv >> 1) * 32;  // wave N offset inside block
  const int mrow = lane & 15;
  const int hgrp = lane >> 4;

  // two 16B chunks per thread per tile: chunk ids t and t+256 of 512
  const int r0 = t >> 2,           c0 = (t & 3) * 8;
  const int r1 = (t + 256) >> 2,   c1 = ((t + 256) & 3) * 8;

  FragC acc[4][2];
  #pragma unroll
  for (int mt = 0; mt < 4; ++mt)
    #pragma unroll
    for (int nt = 0; nt < 2; ++nt)
      #pragma unroll
      for (int r = 0; r < 8; ++r) acc[mt][nt].e[r] = 0.f;

  auto issue_tile = [&](int buf, int kb) {
    async_load_b128(A + (size_t)(bm + r0) * K + kb + c0, &As[buf][r0 * LDST + c0]);
    async_load_b128(A + (size_t)(bm + r1) * K + kb + c1, &As[buf][r1 * LDST + c1]);
    async_load_b128(B + (size_t)(bn + r0) * K + kb + c0, &Bs[buf][r0 * LDST + c0]);
    async_load_b128(B + (size_t)(bn + r1) * K + kb + c1, &Bs[buf][r1 * LDST + c1]);
  };

  // prologue: fill buffer 0
  issue_tile(0, 0);
  wait_async0();
  __syncthreads();

  for (int kb = 0, it = 0; kb < K; kb += BK, ++it) {
    const int cur = it & 1;
    if (kb + BK < K) issue_tile(cur ^ 1, kb + BK);   // overlap next tile

    FragAB af[4], bfg[2];
    #pragma unroll
    for (int mt = 0; mt < 4; ++mt) {
      const __bf16* p = &As[cur][(wm + mt * 16 + mrow) * LDST];
      af[mt].q[0] = *(const uint4*)(p + hgrp * 16);       // K 0..7 / 16..23
      af[mt].q[1] = *(const uint4*)(p + hgrp * 16 + 8);   // K 8..15 / 24..31
    }
    #pragma unroll
    for (int nt = 0; nt < 2; ++nt) {
      const __bf16* p = &Bs[cur][(wn + nt * 16 + mrow) * LDST + hgrp * 16];
      bfg[nt].q[0] = *(const uint4*)(p);
      bfg[nt].q[1] = *(const uint4*)(p + 8);
    }
    #pragma unroll
    for (int mt = 0; mt < 4; ++mt)
      #pragma unroll
      for (int nt = 0; nt < 2; ++nt)
        acc[mt][nt].v = wmma_bf16(af[mt].v, bfg[nt].v, acc[mt][nt].v);

    wait_async0();       // next buffer's async writes landed (this wave)
    __syncthreads();     // ...and every wave's; also fences LDS reads of cur
  }

  // epilogue: C layout -> row = (lane>>4)*8 + r, col = lane&15
  #pragma unroll
  for (int mt = 0; mt < 4; ++mt)
    #pragma unroll
    for (int nt = 0; nt < 2; ++nt)
      #pragma unroll
      for (int r = 0; r < 8; ++r) {
        int row = bm + wm + mt * 16 + hgrp * 8 + r;
        int col = bn + wn + nt * 16 + mrow;
        float x = acc[mt][nt].e[r] + bias[col];
        if (RELU) x = fmaxf(x, 0.f);
        if (OUT_BF16)
          ((__bf16*)Cout)[(size_t)row * N + col] = (__bf16)x;
        else
          ((float*)Cout)[(size_t)row * N + col] = x;
      }
}

// ---------------------------------------------------------------------------
// Fused per-(graph, head, 16-query-tile) attention.
// qkv layout: [T, 3D] bf16, row t = [q | k | v], head h at col h*64.
// Scores strip S[16][512] lives in LDS (f32), softmaxed, converted to bf16
// probs P, then P @ V via WMMA.  128 threads = 4 waves.
// ---------------------------------------------------------------------------
__global__ __launch_bounds__(128)
void attn_kernel(const __bf16* __restrict__ qkv, __bf16* __restrict__ out) {
  constexpr int SSTR = NN + 4;
  __shared__ float  S[16 * SSTR];                               // 33 KB
  __shared__ __attribute__((aligned(16))) __bf16 P[16 * NN];    // 16 KB

  const int bid = blockIdx.x;
  const int qb  = bid & 31;          // 32 query tiles of 16 per (g,h)
  const int hh  = (bid >> 5) & 15;
  const int g   = bid >> 9;

  const size_t base = (size_t)g * NN * TRIPLE;
  const __bf16* Qb = qkv + base + hh * HD;
  const __bf16* Kb = qkv + base + DD + hh * HD;
  const __bf16* Vb = qkv + base + 2 * DD + hh * HD;

  const int t    = threadIdx.x;
  const int lane = t & 31;
  const int wv   = t >> 5;           // 4 waves
  const int mrow = lane & 15;
  const int hgrp = lane >> 4;

  // Q A-fragments for this query tile (hd=64 -> two K-steps of 32)
  FragAB aq[2];
  {
    const __bf16* qr = Qb + (size_t)(qb * 16 + mrow) * TRIPLE;
    #pragma unroll
    for (int kk = 0; kk < 2; ++kk) {
      aq[kk].q[0] = *(const uint4*)(qr + kk * 32 + hgrp * 16);
      aq[kk].q[1] = *(const uint4*)(qr + kk * 32 + hgrp * 16 + 8);
    }
  }

  // ---- scores: 32 key tiles, 8 per wave ----
  #pragma unroll
  for (int j = 0; j < 8; ++j) {
    const int kn = wv + 4 * j;
    FragC c;
    #pragma unroll
    for (int r = 0; r < 8; ++r) c.e[r] = 0.f;
    #pragma unroll
    for (int kk = 0; kk < 2; ++kk) {
      FragAB b;                       // B[k=hd][n=key], contiguous per lane
      const __bf16* kr = Kb + (size_t)(kn * 16 + mrow) * TRIPLE + kk * 32 + hgrp * 16;
      b.q[0] = *(const uint4*)(kr);
      b.q[1] = *(const uint4*)(kr + 8);
      c.v = wmma_bf16(aq[kk].v, b.v, c.v);
    }
    #pragma unroll
    for (int r = 0; r < 8; ++r)
      S[(hgrp * 8 + r) * SSTR + kn * 16 + mrow] = c.e[r] * 0.125f; // 1/sqrt(64)
  }
  __syncthreads();

  // ---- softmax: 16 rows, 4 per wave, wave32 shuffle reductions ----
  #pragma unroll
  for (int rr = 0; rr < 4; ++rr) {
    const int row = wv * 4 + rr;
    float mx = -3.40282e38f;
    for (int c = lane; c < NN; c += 32) mx = fmaxf(mx, S[row * SSTR + c]);
    for (int off = 16; off; off >>= 1) mx = fmaxf(mx, __shfl_xor(mx, off, 32));
    float sum = 0.f;
    for (int c = lane; c < NN; c += 32) {
      float e = __expf(S[row * SSTR + c] - mx);
      S[row * SSTR + c] = e;
      sum += e;
    }
    for (int off = 16; off; off >>= 1) sum += __shfl_xor(sum, off, 32);
    const float inv = 1.f / sum;
    for (int c = lane; c < NN; c += 32)
      P[row * NN + c] = (__bf16)(S[row * SSTR + c] * inv);
  }
  __syncthreads();

  // ---- O = P @ V : 4 hd-tiles of 16, one per wave; K = 512 keys ----
  FragC o;
  #pragma unroll
  for (int r = 0; r < 8; ++r) o.e[r] = 0.f;
  const int nt = wv;
  for (int kc = 0; kc < 16; ++kc) {
    FragAB a, b;
    const __bf16* pr = &P[mrow * NN + kc * 32];
    a.q[0] = *(const uint4*)(pr + hgrp * 16);
    a.q[1] = *(const uint4*)(pr + hgrp * 16 + 8);
    // B[k=key][n=hd]: lane holds hd col (mrow), 16 keys strided by TRIPLE
    #pragma unroll
    for (int e = 0; e < 16; ++e)
      b.e[e] = Vb[(size_t)(kc * 32 + hgrp * 16 + e) * TRIPLE + nt * 16 + mrow];
    o.v = wmma_bf16(a.v, b.v, o.v);
  }
  #pragma unroll
  for (int r = 0; r < 8; ++r) {
    const int q = qb * 16 + hgrp * 8 + r;
    out[(size_t)(g * NN + q) * DD + hh * HD + nt * 16 + mrow] = (__bf16)o.e[r];
  }
}

// ---------------------------------------------------------------------------
// y = LayerNorm(a + c) * g + b ; writes f32 and optional bf16 copy.
// One block (256 threads) per row of D=1024.
// ---------------------------------------------------------------------------
__global__ __launch_bounds__(256)
void add_ln_kernel(const float* __restrict__ A, const float* __restrict__ C,
                   const float* __restrict__ gam, const float* __restrict__ bet,
                   float* __restrict__ outf, __bf16* __restrict__ outb) {
  const int row = blockIdx.x;
  const float* a = A + (size_t)row * DD;
  const float* c = C + (size_t)row * DD;
  const int t = threadIdx.x, lane = t & 31, wv = t >> 5;

  __shared__ float rbuf[8];
  __shared__ float s_mean, s_rstd;

  float x[4];
  float s = 0.f;
  #pragma unroll
  for (int i = 0; i < 4; ++i) {
    x[i] = a[t + 256 * i] + c[t + 256 * i];
    s += x[i];
  }
  for (int off = 16; off; off >>= 1) s += __shfl_xor(s, off, 32);
  if (lane == 0) rbuf[wv] = s;
  __syncthreads();
  if (t == 0) {
    float tot = 0.f;
    for (int i = 0; i < 8; ++i) tot += rbuf[i];
    s_mean = tot * (1.0f / DD);
  }
  __syncthreads();
  const float mu = s_mean;
  float v = 0.f;
  #pragma unroll
  for (int i = 0; i < 4; ++i) { float d = x[i] - mu; v += d * d; }
  for (int off = 16; off; off >>= 1) v += __shfl_xor(v, off, 32);
  if (lane == 0) rbuf[wv] = v;
  __syncthreads();
  if (t == 0) {
    float tot = 0.f;
    for (int i = 0; i < 8; ++i) tot += rbuf[i];
    s_rstd = rsqrtf(tot * (1.0f / DD) + 1e-5f);
  }
  __syncthreads();
  const float rstd = s_rstd;
  #pragma unroll
  for (int i = 0; i < 4; ++i) {
    const int col = t + 256 * i;
    float y = (x[i] - mu) * rstd * gam[col] + bet[col];
    outf[(size_t)row * DD + col] = y;
    if (outb) outb[(size_t)row * DD + col] = (__bf16)y;
  }
}

// ---------------------------------------------------------------------------
// Host-side pipeline
// ---------------------------------------------------------------------------
extern "C" void kernel_launch(void* const* d_in, const int* in_sizes, int n_in,
                              void* d_out, int out_size, void* d_ws, size_t ws_size,
                              hipStream_t stream) {
  (void)in_sizes; (void)n_in; (void)out_size; (void)ws_size;

  const float* h       = (const float*)d_in[0];
  const float* w_in    = (const float*)d_in[1];
  const float* b_in    = (const float*)d_in[2];
  const float* w_out   = (const float*)d_in[3];
  const float* b_out   = (const float*)d_in[4];
  const float* ln1_g   = (const float*)d_in[5];
  const float* ln1_b   = (const float*)d_in[6];
  const float* ln2_g   = (const float*)d_in[7];
  const float* ln2_b   = (const float*)d_in[8];
  const float* ffn_w1  = (const float*)d_in[9];
  const float* ffn_b1  = (const float*)d_in[10];
  const float* ffn_w2  = (const float*)d_in[11];
  const float* ffn_b2  = (const float*)d_in[12];

  // workspace carve-up (bytes, 256B aligned)
  uint8_t* ws = (uint8_t*)d_ws;
  size_t off = 0;
  auto carve = [&](size_t bytes) -> uint8_t* {
    uint8_t* p = ws + off;
    off += (bytes + 255) & ~(size_t)255;
    return p;
  };
  __bf16* hbf    = (__bf16*)carve((size_t)TT * DD * 2);       // 32 MB
  __bf16* winbf  = (__bf16*)carve((size_t)3 * DD * DD * 2);   // 6 MB
  __bf16* woutbf = (__bf16*)carve((size_t)DD * DD * 2);       // 2 MB
  __bf16* w1bf   = (__bf16*)carve((size_t)2 * DD * DD * 2);   // 4 MB
  __bf16* w2bf   = (__bf16*)carve((size_t)DD * 2 * DD * 2);   // 4 MB
  __bf16* qkvbf  = (__bf16*)carve((size_t)TT * TRIPLE * 2);   // 96 MB (reused as f_bf)
  __bf16* attnbf = (__bf16*)carve((size_t)TT * DD * 2);       // 32 MB (reused as x1_bf)
  float*  tmpf   = (float*)carve((size_t)TT * DD * 4);        // 64 MB
  float*  x1f    = (float*)carve((size_t)TT * DD * 4);        // 64 MB
  __bf16* fbf    = qkvbf;                                     // reuse
  __bf16* x1bf   = attnbf;                                    // reuse

  // 1) convert streamed operands to bf16
  {
    int n;
    n = TT * DD;          cvt_f32_bf16<<<(n + 255) / 256, 256, 0, stream>>>(h,      hbf,    n);
    n = 3 * DD * DD;      cvt_f32_bf16<<<(n + 255) / 256, 256, 0, stream>>>(w_in,   winbf,  n);
    n = DD * DD;          cvt_f32_bf16<<<(n + 255) / 256, 256, 0, stream>>>(w_out,  woutbf, n);
    n = 2 * DD * DD;      cvt_f32_bf16<<<(n + 255) / 256, 256, 0, stream>>>(ffn_w1, w1bf,   n);
    n = DD * 2 * DD;      cvt_f32_bf16<<<(n + 255) / 256, 256, 0, stream>>>(ffn_w2, w2bf,   n);
  }

  // 2) qkv = h @ w_in^T + b_in   [T, 3D] bf16
  gemm_bf16_kernel<true, false><<<dim3(3 * DD / 128, TT / 128), 256, 0, stream>>>(
      hbf, winbf, b_in, qkvbf, TT, 3 * DD, DD);

  // 3) fused attention -> attn_out [T, D] bf16
  attn_kernel<<<GG * HH * (NN / 16), 128, 0, stream>>>(qkvbf, attnbf);

  // 4) o = attn_out @ w_out^T + b_out  -> tmp f32
  gemm_bf16_kernel<false, false><<<dim3(DD / 128, TT / 128), 256, 0, stream>>>(
      attnbf, woutbf, b_out, tmpf, TT, DD, DD);

  // 5) x1 = LN(h + o)  -> x1f (f32) + x1bf (bf16, overwrites attn buffer)
  add_ln_kernel<<<TT, 256, 0, stream>>>(h, tmpf, ln1_g, ln1_b, x1f, x1bf);

  // 6) f = relu(x1 @ w1^T + b1)  [T, 2D] bf16 (reuses qkv buffer)
  gemm_bf16_kernel<true, true><<<dim3(2 * DD / 128, TT / 128), 256, 0, stream>>>(
      x1bf, w1bf, ffn_b1, fbf, TT, 2 * DD, DD);

  // 7) f2 = f @ w2^T + b2  -> tmp f32
  gemm_bf16_kernel<false, false><<<dim3(DD / 128, TT / 128), 256, 0, stream>>>(
      fbf, w2bf, ffn_b2, tmpf, TT, DD, 2 * DD);

  // 8) out = LN(x1 + f2)  -> d_out f32
  add_ln_kernel<<<TT, 256, 0, stream>>>(x1f, tmpf, ln2_g, ln2_b, (float*)d_out, nullptr);
}
